// LoRAAttnProcessor_8220567405163
// MI455X (gfx1250) — matrix operand order repaired and checked by reading, not compiled, not run on hardware
//
#include <hip/hip_runtime.h>
#include <hip/hip_bf16.h>

// ---------------------------------------------------------------------------
// Types for CDNA5 WMMA (wave32): v_wmma_f32_16x16x32_bf16
// ---------------------------------------------------------------------------
typedef __attribute__((ext_vector_type(16))) __bf16 bf16x16;
typedef __attribute__((ext_vector_type(8)))  __bf16 bf16x8;
typedef __attribute__((ext_vector_type(8)))  float  f32x8;

#define D_MODEL 1280
#define HEADS   20
#define HDIM    64
#define SEQ     1024
#define BATCH   8
#define MTOT    (BATCH * SEQ)      // 8192
#define BH      (BATCH * HEADS)    // 160

__device__ __forceinline__ bf16x16 concat8(bf16x8 lo, bf16x8 hi) {
    return __builtin_shufflevector(lo, hi, 0,1,2,3,4,5,6,7,8,9,10,11,12,13,14,15);
}

// A-matrix 16x32 bf16 fragment: lane L: row passed by caller;
// K elems: k0 + 16*(L>>4) + e, contiguous. One 32B load.
__device__ __forceinline__ bf16x16 load_a_frag(const __bf16* base, long row, int ld,
                                               int k0, int lane) {
    const __bf16* p = base + row * (long)ld + k0 + 16 * (lane >> 4);
    return *(const bf16x16*)p;
}

// B-matrix 32x16 bf16 fragment: lane L: col passed by caller;
// K elems: k0 + 16*(L>>4) + e, contiguous. One 32B load.
__device__ __forceinline__ bf16x16 load_b_frag(const __bf16* base, long col, int ld,
                                               int k0, int lane) {
    const __bf16* p = base + col * (long)ld + k0 + 16 * (lane >> 4);
    return *(const bf16x16*)p;
}

// ---------------------------------------------------------------------------
// Prep kernels
// ---------------------------------------------------------------------------
__global__ void cvt_x_kernel(const float* __restrict__ x, __bf16* __restrict__ xb, int n) {
    int i = blockIdx.x * 256 + threadIdx.x;
    if (i < n) xb[i] = (__bf16)x[i];
}

// W_eff = W + up @ down  (LoRA fold, rank 4), to bf16.
__global__ void fold_w_kernel(const float* __restrict__ W, const float* __restrict__ down,
                              const float* __restrict__ up, __bf16* __restrict__ out) {
    int i = blockIdx.x * 256 + threadIdx.x;        // i < 1280*1280
    int n = i / D_MODEL, kk = i % D_MODEL;
    float a = W[i];
#pragma unroll
    for (int r = 0; r < 4; ++r) a += up[n * 4 + r] * down[r * D_MODEL + kk];
    out[i] = (__bf16)a;
}

// ---------------------------------------------------------------------------
// Fused QKV GEMM: C[8192,1280] = Xbf @ W_eff^T, epilogue scatters to
// attention layouts. blockIdx.z selects q/k/v. q scaled by 0.125 (1/sqrt(64)).
// Block = 256 threads = 8 waves; each wave owns a 64x64 tile -> 256x128/block.
// ---------------------------------------------------------------------------
__global__ __launch_bounds__(256) void qkv_gemm_kernel(
    const __bf16* __restrict__ x,
    const __bf16* __restrict__ wq, const __bf16* __restrict__ wk, const __bf16* __restrict__ wv,
    __bf16* __restrict__ qout, __bf16* __restrict__ kout, __bf16* __restrict__ vtout) {

    const int lane = threadIdx.x & 31;
    const int wave = threadIdx.x >> 5;
    const int hf   = lane >> 4;
    const int nl   = lane & 15;
    const int m0   = blockIdx.x * 128 + (wave >> 1) * 64;
    const int n0   = blockIdx.y * 128 + (wave & 1) * 64;
    const int z    = blockIdx.z;
    const __bf16* w = (z == 0) ? wq : (z == 1) ? wk : wv;

    f32x8 acc[4][4];
    const f32x8 fz = {0.f,0.f,0.f,0.f,0.f,0.f,0.f,0.f};
#pragma unroll
    for (int mi = 0; mi < 4; ++mi)
#pragma unroll
        for (int ni = 0; ni < 4; ++ni) acc[mi][ni] = fz;

    for (int k0 = 0; k0 < D_MODEL; k0 += 32) {
        bf16x16 fa[4], fb[4];
#pragma unroll
        for (int i = 0; i < 4; ++i)
            fa[i] = load_a_frag(x, (long)(m0 + i * 16 + nl), D_MODEL, k0, lane);
#pragma unroll
        for (int i = 0; i < 4; ++i)
            fb[i] = load_b_frag(w, (long)(n0 + i * 16 + nl), D_MODEL, k0, lane);
#pragma unroll
        for (int mi = 0; mi < 4; ++mi)
#pragma unroll
            for (int ni = 0; ni < 4; ++ni)
                acc[mi][ni] = __builtin_amdgcn_wmma_f32_16x16x32_bf16(
                    false, fa[mi], false, fb[ni], (short)0, acc[mi][ni], false, false);
    }

    // epilogue: a 64-row M-tile never crosses a batch (m0 is a multiple of 64)
    const int bb = m0 >> 10;
#pragma unroll
    for (int mi = 0; mi < 4; ++mi)
#pragma unroll
        for (int ni = 0; ni < 4; ++ni) {
            const int n  = n0 + ni * 16 + nl;
            const int h  = n >> 6, hd = n & 63;
            const int s0 = (m0 + mi * 16 + 8 * hf) & 1023;   // 8 consecutive s per lane
            if (z == 2) {
                // V stored transposed [bh][hd][S]: 8 rows are contiguous in s
                bf16x8 pk;
#pragma unroll
                for (int r = 0; r < 8; ++r) pk[r] = (__bf16)acc[mi][ni][r];
                *(bf16x8*)(vtout + ((long)(bb * HEADS + h) * HDIM + hd) * SEQ + s0) = pk;
            } else {
                __bf16* dst = (z == 0) ? qout : kout;
                const float sc = (z == 0) ? 0.125f : 1.0f;
#pragma unroll
                for (int r = 0; r < 8; ++r)
                    dst[((long)(bb * HEADS + h) * SEQ + (s0 + r)) * HDIM + hd] =
                        (__bf16)(acc[mi][ni][r] * sc);
            }
        }
}

// ---------------------------------------------------------------------------
// Flash attention: one wave owns 32 query rows; block = 4 waves = 128 q rows.
// grid = (1024/128, B*H). Online softmax; P transposed C->A via per-wave LDS.
// ---------------------------------------------------------------------------
__global__ __launch_bounds__(128) void attn_kernel(
    const __bf16* __restrict__ q, const __bf16* __restrict__ k,
    const __bf16* __restrict__ vt, __bf16* __restrict__ ctx) {

    __shared__ __bf16 ldsP[4][32 * 32];   // 2KB per wave

    const int lane = threadIdx.x & 31;
    const int wave = threadIdx.x >> 5;
    const int hf   = lane >> 4;
    const int nl   = lane & 15;
    const int bh   = blockIdx.y;
    const int bb   = bh / HEADS, h = bh % HEADS;
    const int qBase = blockIdx.x * 128 + wave * 32;

    const __bf16* qp = q  + (long)bh * SEQ * HDIM;
    const __bf16* kp = k  + (long)bh * SEQ * HDIM;
    const __bf16* vp = vt + (long)bh * HDIM * SEQ;

    // Q fragments (held for the whole loop); q already scaled by 1/sqrt(hd)
    bf16x16 qf[2][2];
#pragma unroll
    for (int mi = 0; mi < 2; ++mi)
#pragma unroll
        for (int ks = 0; ks < 2; ++ks)
            qf[mi][ks] = load_a_frag(qp, (long)(qBase + mi * 16 + nl), HDIM, ks * 32, lane);

    f32x8 o[2][4];
    const f32x8 fz = {0.f,0.f,0.f,0.f,0.f,0.f,0.f,0.f};
#pragma unroll
    for (int mi = 0; mi < 2; ++mi)
#pragma unroll
        for (int ni = 0; ni < 4; ++ni) o[mi][ni] = fz;

    float mrow[16], lrow[16];
#pragma unroll
    for (int i = 0; i < 16; ++i) { mrow[i] = -3.0e38f; lrow[i] = 0.f; }

    for (int kt = 0; kt < SEQ; kt += 32) {
        // ---- S = Q @ K^T over this 32-key chunk -------------------------
        bf16x16 kf[2][2];
#pragma unroll
        for (int ni = 0; ni < 2; ++ni)
#pragma unroll
            for (int ks = 0; ks < 2; ++ks)
                kf[ni][ks] = load_b_frag(kp, (long)(kt + ni * 16 + nl), HDIM, ks * 32, lane);

        f32x8 sc[2][2];
#pragma unroll
        for (int mi = 0; mi < 2; ++mi)
#pragma unroll
            for (int ni = 0; ni < 2; ++ni) {
                sc[mi][ni] = fz;
#pragma unroll
                for (int ks = 0; ks < 2; ++ks)
                    sc[mi][ni] = __builtin_amdgcn_wmma_f32_16x16x32_bf16(
                        false, qf[mi][ks], false, kf[ni][ks], (short)0, sc[mi][ni], false, false);
            }

        // ---- online softmax (rows live in 16-lane halves; xor<=8 stays put)
#pragma unroll
        for (int mi = 0; mi < 2; ++mi)
#pragma unroll
            for (int r = 0; r < 8; ++r) {
                const int idx = mi * 8 + r;
                float v0 = sc[mi][0][r], v1 = sc[mi][1][r];
                float rmax = fmaxf(v0, v1);
#pragma unroll
                for (int off = 1; off < 16; off <<= 1)
                    rmax = fmaxf(rmax, __shfl_xor(rmax, off, 32));
                const float mnew  = fmaxf(mrow[idx], rmax);
                const float alpha = __expf(mrow[idx] - mnew);
                const float p0 = __expf(v0 - mnew);
                const float p1 = __expf(v1 - mnew);
                sc[mi][0][r] = p0; sc[mi][1][r] = p1;
                float rsum = p0 + p1;
#pragma unroll
                for (int off = 1; off < 16; off <<= 1)
                    rsum += __shfl_xor(rsum, off, 32);
                lrow[idx] = lrow[idx] * alpha + rsum;
                mrow[idx] = mnew;
#pragma unroll
                for (int ni = 0; ni < 4; ++ni) o[mi][ni][r] *= alpha;
            }

        // ---- transpose P (C-layout -> A-layout) via per-wave LDS tile ---
        __syncthreads();
#pragma unroll
        for (int mi = 0; mi < 2; ++mi)
#pragma unroll
            for (int ni = 0; ni < 2; ++ni)
#pragma unroll
                for (int r = 0; r < 8; ++r)
                    ldsP[wave][(mi * 16 + 8 * hf + r) * 32 + ni * 16 + nl] =
                        (__bf16)sc[mi][ni][r];
        __syncthreads();

        bf16x16 pf[2];
#pragma unroll
        for (int mi = 0; mi < 2; ++mi)
            pf[mi] = load_a_frag(&ldsP[wave][0], (long)(mi * 16 + nl), 32, 0, lane);

        // ---- O += P @ V (V transposed: contiguous over keys per lane) ---
        bf16x16 vf[4];
#pragma unroll
        for (int ni = 0; ni < 4; ++ni)
            vf[ni] = load_b_frag(vp, (long)(ni * 16 + nl), SEQ, kt, lane);
#pragma unroll
        for (int mi = 0; mi < 2; ++mi)
#pragma unroll
            for (int ni = 0; ni < 4; ++ni)
                o[mi][ni] = __builtin_amdgcn_wmma_f32_16x16x32_bf16(
                    false, pf[mi], false, vf[ni], (short)0, o[mi][ni], false, false);
    }

    // ---- normalize and write ctx [m, 1280] bf16 -------------------------
#pragma unroll
    for (int mi = 0; mi < 2; ++mi)
#pragma unroll
        for (int ni = 0; ni < 4; ++ni)
#pragma unroll
            for (int r = 0; r < 8; ++r) {
                const float v = o[mi][ni][r] / lrow[mi * 8 + r];
                const int sQ = qBase + mi * 16 + 8 * hf + r;
                ctx[((long)(bb * SEQ + sQ)) * D_MODEL + h * HDIM + ni * 16 + nl] = (__bf16)v;
            }
}

// ---------------------------------------------------------------------------
// Output projection: out = ctx @ Wo_eff^T + bo, f32 output.
// ---------------------------------------------------------------------------
__global__ __launch_bounds__(256) void o_gemm_kernel(
    const __bf16* __restrict__ ctx, const __bf16* __restrict__ wo,
    const float* __restrict__ bo, float* __restrict__ out) {

    const int lane = threadIdx.x & 31;
    const int wave = threadIdx.x >> 5;
    const int hf   = lane >> 4;
    const int nl   = lane & 15;
    const int m0   = blockIdx.x * 128 + (wave >> 1) * 64;
    const int n0   = blockIdx.y * 128 + (wave & 1) * 64;

    f32x8 acc[4][4];
    const f32x8 fz = {0.f,0.f,0.f,0.f,0.f,0.f,0.f,0.f};
#pragma unroll
    for (int mi = 0; mi < 4; ++mi)
#pragma unroll
        for (int ni = 0; ni < 4; ++ni) acc[mi][ni] = fz;

    for (int k0 = 0; k0 < D_MODEL; k0 += 32) {
        bf16x16 fa[4], fb[4];
#pragma unroll
        for (int i = 0; i < 4; ++i)
            fa[i] = load_a_frag(ctx, (long)(m0 + i * 16 + nl), D_MODEL, k0, lane);
#pragma unroll
        for (int i = 0; i < 4; ++i)
            fb[i] = load_b_frag(wo, (long)(n0 + i * 16 + nl), D_MODEL, k0, lane);
#pragma unroll
        for (int mi = 0; mi < 4; ++mi)
#pragma unroll
            for (int ni = 0; ni < 4; ++ni)
                acc[mi][ni] = __builtin_amdgcn_wmma_f32_16x16x32_bf16(
                    false, fa[mi], false, fb[ni], (short)0, acc[mi][ni], false, false);
    }

#pragma unroll
    for (int mi = 0; mi < 4; ++mi)
#pragma unroll
        for (int ni = 0; ni < 4; ++ni) {
            const int n = n0 + ni * 16 + nl;
            const float bias = bo[n];
#pragma unroll
            for (int r = 0; r < 8; ++r) {
                const int m = m0 + mi * 16 + 8 * hf + r;
                out[(long)m * D_MODEL + n] = acc[mi][ni][r] + bias;
            }
        }
}

// ---------------------------------------------------------------------------
// Host launcher
// ---------------------------------------------------------------------------
extern "C" void kernel_launch(void* const* d_in, const int* in_sizes, int n_in,
                              void* d_out, int out_size, void* d_ws, size_t ws_size,
                              hipStream_t stream) {
    (void)in_sizes; (void)n_in; (void)out_size; (void)ws_size;

    const float* x  = (const float*)d_in[0];
    const float* Wq = (const float*)d_in[1];
    const float* Wk = (const float*)d_in[2];
    const float* Wv = (const float*)d_in[3];
    const float* Wo = (const float*)d_in[4];
    const float* bo = (const float*)d_in[5];
    const float* qd = (const float*)d_in[6];
    const float* qu = (const float*)d_in[7];
    const float* kd = (const float*)d_in[8];
    const float* ku = (const float*)d_in[9];
    const float* vd = (const float*)d_in[10];
    const float* vu = (const float*)d_in[11];
    const float* od = (const float*)d_in[12];
    const float* ou = (const float*)d_in[13];

    // workspace carve-up (bf16 elements)
    __bf16* xb  = (__bf16*)d_ws;                       // 8192*1280
    __bf16* wq  = xb  + (long)MTOT * D_MODEL;          // 1280*1280 each
    __bf16* wk  = wq  + (long)D_MODEL * D_MODEL;
    __bf16* wv  = wk  + (long)D_MODEL * D_MODEL;
    __bf16* wo  = wv  + (long)D_MODEL * D_MODEL;
    __bf16* qb  = wo  + (long)D_MODEL * D_MODEL;       // [BH][S][hd]
    __bf16* kb  = qb  + (long)BH * SEQ * HDIM;         // [BH][S][hd]
    __bf16* vtb = kb  + (long)BH * SEQ * HDIM;         // [BH][hd][S]
    __bf16* ctx = vtb + (long)BH * HDIM * SEQ;         // [M][D]

    const int nx = MTOT * D_MODEL;                     // 10,485,760
    cvt_x_kernel<<<(nx + 255) / 256, 256, 0, stream>>>(x, xb, nx);

    const int nw = D_MODEL * D_MODEL;                  // 1,638,400
    fold_w_kernel<<<nw / 256, 256, 0, stream>>>(Wq, qd, qu, wq);
    fold_w_kernel<<<nw / 256, 256, 0, stream>>>(Wk, kd, ku, wk);
    fold_w_kernel<<<nw / 256, 256, 0, stream>>>(Wv, vd, vu, wv);
    fold_w_kernel<<<nw / 256, 256, 0, stream>>>(Wo, od, ou, wo);

    qkv_gemm_kernel<<<dim3(MTOT / 128, D_MODEL / 128, 3), 256, 0, stream>>>(
        xb, wq, wk, wv, qb, kb, vtb);

    attn_kernel<<<dim3(SEQ / 128, BH), 128, 0, stream>>>(qb, kb, vtb, ctx);

    o_gemm_kernel<<<dim3(MTOT / 128, D_MODEL / 128), 256, 0, stream>>>(
        ctx, wo, bo, (float*)d_out);
}
